// SiameseAlexNet_3642132267725
// MI455X (gfx1250) — matrix operand order repaired and checked by reading, not compiled, not run on hardware
//
#include <hip/hip_runtime.h>
#include <hip/hip_bf16.h>
#include <cstdint>
#include <cstddef>

// ---------------------------------------------------------------------------
// B=128, Cin=512 (concat), Cout=256, H=W=24; 3x3 pad-1 conv -> BN -> ReLU ->
// 1x1 conv to 25 -> channel softmax -> 5x5 weighted shift-propagation.
// Conv1 = implicit GEMM M=256, N=73728, K=4608 on v_wmma_f32_16x16x32_f16,
// K visited tap-major (9 taps x 512 ci): im2col gather = 1 base + immediate
// offsets, no div/mod in the hot loop.  128x128 tile -> 8 WMMA / wave / step.
// ---------------------------------------------------------------------------
#define B_N   128
#define C_IN  512
#define C_OUT 256
#define HH    24
#define WW    24
#define HP    26
#define WP    26
#define PLANE (HP*WP)       // 676
#define KTOT  4608          // 9*512, tap-major
#define NTOT  (B_N*HH*WW)   // 73728
#define KSTEPS (KTOT/32)    // 144

typedef __attribute__((ext_vector_type(16))) _Float16 v16h;
typedef __attribute__((ext_vector_type(8)))  float    v8f;

union Half8  { _Float16 h[8]; uint4 u; };
union Frag16 { uint4 u[2];    v16h  v; };

// ---------------------------------------------------------------------------
// Kernel 1: concat + fp32->fp16 + spatial zero-pad into xh[b][ci][yy][xx]
// ---------------------------------------------------------------------------
__global__ __launch_bounds__(256) void k_pad_f16(
    const float* __restrict__ src, const float* __restrict__ det,
    _Float16* __restrict__ xh) {
  size_t idx = (size_t)blockIdx.x * 256 + threadIdx.x;
  if (idx >= (size_t)B_N * C_IN * PLANE) return;
  int xx = (int)(idx % WP); size_t t = idx / WP;
  int yy = (int)(t % HP);   t /= HP;
  int ci = (int)(t % C_IN); int b = (int)(t / C_IN);
  float v = 0.f;
  if (yy >= 1 && yy <= HH && xx >= 1 && xx <= WW) {
    const float* p = (ci < 256) ? src : det;
    int c = ci & 255;
    v = p[(((size_t)b * 256 + c) * HH + (yy - 1)) * WW + (xx - 1)];
  }
  xh[idx] = (_Float16)v;
}

// ---------------------------------------------------------------------------
// Kernel 2: repack w1 (OIHW f32) -> w1h[co][kidx] f16, TAP-MAJOR:
//   kidx = tap*512 + ci,  tap = kh*3+kw
// ---------------------------------------------------------------------------
__global__ __launch_bounds__(256) void k_pack_w1(
    const float* __restrict__ w1, _Float16* __restrict__ w1h) {
  int idx = blockIdx.x * 256 + threadIdx.x;
  if (idx >= C_OUT * KTOT) return;
  int kidx = idx % KTOT, co = idx / KTOT;
  int tap = kidx / C_IN, ci = kidx % C_IN;
  int kh = tap / 3, kw = tap % 3;
  w1h[idx] = (_Float16)w1[(((size_t)co * C_IN + ci) * 3 + kh) * 3 + kw];
}

// ---------------------------------------------------------------------------
// Kernel 3: conv1 implicit GEMM.  128(M) x 128(N) tile per 256-thread WG,
// K-step 32, double-buffered LDS (32 KB).  8 waves: wave w -> M sub-tiles
// {2*(w&3), 2*(w&3)+1} x N sub-tiles [(w>>2)*4 .. +3] -> 8 WMMA per K-step,
// each B fragment feeding two independent accumulators.
// Fused bias+BN+ReLU epilogue -> yT[n][co] f16.
// ---------------------------------------------------------------------------
__global__ __launch_bounds__(256) void k_conv1_wmma(
    const _Float16* __restrict__ xh, const _Float16* __restrict__ w1h,
    const float* __restrict__ b1,  const float* __restrict__ gam,
    const float* __restrict__ bet, const float* __restrict__ mea,
    const float* __restrict__ var, _Float16* __restrict__ yT) {
  __shared__ __align__(16) _Float16 As[2][128][32];  // [m][k]  16 KB
  __shared__ __align__(16) _Float16 Bs[2][128][32];  // [n][k]  16 KB

  const int tid  = threadIdx.x;
  const int lane = tid & 31;
  const int wave = tid >> 5;
  const int nblk = blockIdx.x * 128;  // spatial block
  const int mblk = blockIdx.y * 128;  // out-channel block

  // ---- A fill: 4096 halves -> 16/thread (two b128 load+store)
  const int fm = tid >> 1;            // 0..127
  const int fkA = (tid & 1) * 16;     // 0 or 16
  const _Float16* aRow = w1h + (size_t)(mblk + fm) * KTOT + fkA;

  // ---- B fill: 4096 halves -> 16/thread (16 u16 loads, two b128 stores)
  const int fc = (tid & 1) * 16;      // ci run start within the 32-k step
  const int n_f = nblk + fm;
  const int b_f = n_f / (HH * WW);
  const int r_f = n_f % (HH * WW);
  const int y_f = r_f / WW, x_f = r_f % WW;
  const _Float16* xrow = xh + (size_t)b_f * C_IN * PLANE + y_f * WP + x_f;

  auto fill = [&](int buf, int kt) {
    const int k0 = kt * 32;
    // A tile (contiguous tap-major k)
    *(uint4*)&As[buf][fm][fkA]     = *(const uint4*)&aRow[k0];
    *(uint4*)&As[buf][fm][fkA + 8] = *(const uint4*)&aRow[k0 + 8];
    // B tile: tap = kt>>4, ci base = (kt&15)*32; no div/mod (tap<=8 trick)
    const int tap = kt >> 4;
    const int cb  = (kt & 15) * 32;
    const int kh  = (tap * 11) >> 5;  // tap/3 for tap in [0,8]
    const int kw  = tap - 3 * kh;
    const _Float16* p = xrow + (size_t)(cb + fc) * PLANE + kh * WP + kw;
    __builtin_prefetch(p + 32 * PLANE, 0, 1);   // global_prefetch_b8: next ci block
    Half8 u0, u1;
#pragma unroll
    for (int j = 0; j < 8; ++j) u0.h[j] = p[j * PLANE];
#pragma unroll
    for (int j = 0; j < 8; ++j) u1.h[j] = p[(8 + j) * PLANE];
    *(uint4*)&Bs[buf][fm][fc]     = u0.u;
    *(uint4*)&Bs[buf][fm][fc + 8] = u1.u;
  };

  // ---- per-wave WMMA tile assignment
  const int mi2 = (wave & 3) * 2;         // first of 2 M sub-tiles
  const int ng  = (wave >> 2) * 4;        // first of 4 N sub-tiles
  const int lmA = mi2 * 16 + (lane & 15); // A row (sub-tile 0); +16 for sub-tile 1
  const int kbA = (lane >> 4) * 8;        // ISA 16-bit A fragment layout
  const int kbB = (lane >> 4) * 8;        // ISA 16-bit B fragment layout

  v8f acc[2][4];
#pragma unroll
  for (int u = 0; u < 2; ++u)
#pragma unroll
    for (int t = 0; t < 4; ++t) acc[u][t] = (v8f){};

  fill(0, 0);
  __syncthreads();
  int buf = 0;
#pragma unroll 1
  for (int kt = 0; kt < KSTEPS; ++kt) {
    const int nxt = buf ^ 1;
    if (kt + 1 < KSTEPS) fill(nxt, kt + 1);

    Frag16 a0, a1;
    a0.u[0] = *(const uint4*)&As[buf][lmA][kbA];
    a0.u[1] = *(const uint4*)&As[buf][lmA][kbA + 16];
    a1.u[0] = *(const uint4*)&As[buf][lmA + 16][kbA];
    a1.u[1] = *(const uint4*)&As[buf][lmA + 16][kbA + 16];
#pragma unroll
    for (int t = 0; t < 4; ++t) {
      Frag16 bf;
      const int ln = (ng + t) * 16 + (lane & 15);
      bf.u[0] = *(const uint4*)&Bs[buf][ln][kbB];
      bf.u[1] = *(const uint4*)&Bs[buf][ln][kbB + 16];
      acc[0][t] = __builtin_amdgcn_wmma_f32_16x16x32_f16(
          false, a0.v, false, bf.v, (short)0, acc[0][t], false, false);
      acc[1][t] = __builtin_amdgcn_wmma_f32_16x16x32_f16(
          false, a1.v, false, bf.v, (short)0, acc[1][t], false, false);
    }
    __syncthreads();
    buf = nxt;
  }

  // ---- fused bias + BN + ReLU; D layout: vgpr r -> m = r + 8*(lane>=16)
  const int mhi = (lane >> 4) * 8;
#pragma unroll
  for (int u = 0; u < 2; ++u) {
#pragma unroll
    for (int r = 0; r < 8; ++r) {
      const int c = mblk + (mi2 + u) * 16 + mhi + r;
      const float inv = gam[c] * rsqrtf(var[c] + 1e-5f);
      const float sh  = bet[c] - mea[c] * inv;
      const float bb  = b1[c];
#pragma unroll
      for (int t = 0; t < 4; ++t) {
        const int n0 = nblk + (ng + t) * 16 + (lane & 15);
        const float v = fmaxf((acc[u][t][r] + bb) * inv + sh, 0.f);
        yT[(size_t)n0 * C_OUT + c] = (_Float16)v;
      }
    }
  }
}

// ---------------------------------------------------------------------------
// Kernel 4: 1x1 conv 256->25 + channel softmax.  One thread per (b,y,x).
// ---------------------------------------------------------------------------
__global__ __launch_bounds__(256) void k_conv2_softmax(
    const _Float16* __restrict__ yT, const float* __restrict__ w2,
    const float* __restrict__ b2, float* __restrict__ kmaps) {
  __shared__ float w2s[25 * 256];
  for (int i = threadIdx.x; i < 25 * 256; i += 256) w2s[i] = w2[i];
  __syncthreads();

  const int n = blockIdx.x * 256 + threadIdx.x;  // NTOT = 288*256 exactly
  const _Float16* yrow = yT + (size_t)n * C_OUT;

  float acc[25];
#pragma unroll
  for (int o = 0; o < 25; ++o) acc[o] = b2[o];

  for (int ch = 0; ch < C_OUT / 8; ++ch) {
    Half8 h8; h8.u = *(const uint4*)&yrow[ch * 8];
    float f[8];
#pragma unroll
    for (int j = 0; j < 8; ++j) f[j] = (float)h8.h[j];
#pragma unroll
    for (int o = 0; o < 25; ++o) {
      float s = 0.f;
#pragma unroll
      for (int j = 0; j < 8; ++j) s += f[j] * w2s[o * 256 + ch * 8 + j];
      acc[o] += s;
    }
  }

  float mx = acc[0];
#pragma unroll
  for (int o = 1; o < 25; ++o) mx = fmaxf(mx, acc[o]);
  float sum = 0.f;
#pragma unroll
  for (int o = 0; o < 25; ++o) { acc[o] = __expf(acc[o] - mx); sum += acc[o]; }
  const float rs = 1.f / sum;
  float* kp = kmaps + (size_t)n * 25;
#pragma unroll
  for (int o = 0; o < 25; ++o) kp[o] = acc[o] * rs;
}

// ---------------------------------------------------------------------------
// Kernel 5: 5x5 weighted shift-propagation.  One WG per (b,y); one thread per
// channel; the row's 24x25 kmap block staged in LDS.
// ---------------------------------------------------------------------------
__global__ __launch_bounds__(256) void k_propagate(
    const float* __restrict__ src, const float* __restrict__ kmaps,
    float* __restrict__ out) {
  __shared__ float kms[WW * 25];
  const int by = blockIdx.x;           // b*24 + y
  const int b = by / HH, y = by % HH;
  const int nb0 = by * WW;
  for (int i = threadIdx.x; i < WW * 25; i += 256)
    kms[i] = kmaps[(size_t)nb0 * 25 + i];
  __syncthreads();

  const int c = threadIdx.x;
  const float* sc = src + (size_t)(b * C_OUT + c) * HH * WW;
  float* op = out + ((size_t)(b * C_OUT + c) * HH + y) * WW;
  for (int x = 0; x < WW; ++x) {
    float a = 0.f;
#pragma unroll
    for (int o = 0; o < 25; ++o) {           // DELTA: dy outer, dx inner
      const int dy = o / 5 - 2, dx = o % 5 - 2;
      const int sy = y + dy, sx = x + dx;
      if (sy >= 0 && sy < HH && sx >= 0 && sx < WW)
        a += kms[x * 25 + o] * sc[sy * WW + sx];
    }
    op[x] = a;
  }
}

// ---------------------------------------------------------------------------
extern "C" void kernel_launch(void* const* d_in, const int* in_sizes, int n_in,
                              void* d_out, int out_size, void* d_ws, size_t ws_size,
                              hipStream_t stream) {
  const float* src = (const float*)d_in[0];
  const float* det = (const float*)d_in[1];
  const float* w1  = (const float*)d_in[2];
  const float* b1  = (const float*)d_in[3];
  const float* gam = (const float*)d_in[4];
  const float* bet = (const float*)d_in[5];
  const float* mea = (const float*)d_in[6];
  const float* var = (const float*)d_in[7];
  const float* w2  = (const float*)d_in[8];
  const float* b2  = (const float*)d_in[9];
  float* out = (float*)d_out;

  char* ws = (char*)d_ws;
  size_t off = 0;
  _Float16* xh  = (_Float16*)(ws + off); off += (size_t)B_N * C_IN * PLANE * 2; // 88.6 MB
  _Float16* w1h = (_Float16*)(ws + off); off += (size_t)C_OUT * KTOT * 2;       //  2.4 MB
  _Float16* yT  = (_Float16*)(ws + off); off += (size_t)NTOT * C_OUT * 2;       // 37.7 MB
  float*    km  = (float*)(ws + off);    off += (size_t)NTOT * 25 * 4;          //  7.4 MB

  {
    const int total = B_N * C_IN * PLANE;
    k_pad_f16<<<(total + 255) / 256, 256, 0, stream>>>(src, det, xh);
  }
  k_pack_w1<<<(C_OUT * KTOT + 255) / 256, 256, 0, stream>>>(w1, w1h);

  dim3 g1(NTOT / 128, C_OUT / 128);   // 576 x 2 workgroups
  k_conv1_wmma<<<g1, 256, 0, stream>>>(xh, w1h, b1, gam, bet, mea, var, yT);

  k_conv2_softmax<<<NTOT / 256, 256, 0, stream>>>(yT, w2, b2, km);
  k_propagate<<<B_N * HH, 256, 0, stream>>>(src, km, out);
}
